// MergingModel_90890097918088
// MI455X (gfx1250) — matrix operand order repaired and checked by reading, not compiled, not run on hardware
//
#include <hip/hip_runtime.h>
#include <hip/hip_bf16.h>
#include <math.h>

#define N_OBS    1000000
#define N_IMAGES 8192
#define N_ASU    131072
#define MC       32
#define N_OPS    4
#define LOG_2PI  1.8378770664093453f

typedef __attribute__((ext_vector_type(16))) _Float16 v16h;
typedef __attribute__((ext_vector_type(8)))  float    v8f;

// branchless softplus on HW transcendentals: max(x,0) + log(1+exp(-|x|))
__device__ __forceinline__ float softplus_f(float x) {
    return fmaxf(x, 0.0f) + __logf(1.0f + __expf(-fabsf(x)));
}

__device__ __forceinline__ float wave_sum32(float v) {
#pragma unroll
    for (int m = 16; m >= 1; m >>= 1) v += __shfl_xor(v, m, 32);
    return v;
}

// ---------------------------------------------------------------------------
// k_zero: reset per-launch accumulators (ll_img segment sums + f64 scalars)
// ---------------------------------------------------------------------------
__global__ void k_zero(float* ll_img, double* acc) {
    int t = blockIdx.x * blockDim.x + threadIdx.x;
    if (t < N_IMAGES * N_OPS) ll_img[t] = 0.0f;
    if (t < 8) acc[t] = 0.0;
}

// ---------------------------------------------------------------------------
// k_prep: sigma = softplus(raw_scale); KL reduction; per-ASU moments of
// z = loc + sigma*eps over the MC axis:  zstat = (E[z], E[z^2]).
// The MC mean of the Gaussian LL is an exact function of these moments, so
// the main kernel never touches eps (kills the 16MB gather table entirely).
// Tables as_tab/zstat are 1MB each -> resident in the 192MB L2.
// ---------------------------------------------------------------------------
__global__ void k_prep(const float* __restrict__ loc,
                       const float* __restrict__ raw_scale,
                       const float* __restrict__ eps,
                       float2* __restrict__ as_tab,   // (loc, sigma)
                       float2* __restrict__ zstat,    // (E[z], E[z^2])
                       double* acc) {
    int a = blockIdx.x * blockDim.x + threadIdx.x;   // N_ASU divisible by 256
    float s = softplus_f(raw_scale[a]);
    float l = loc[a];
    as_tab[a] = make_float2(l, s);
    float zs = 0.0f, zq = 0.0f;
#pragma unroll
    for (int mc = 0; mc < MC; ++mc) {
        float z = fmaf(s, eps[(size_t)mc * N_ASU + a], l);
        zs += z;
        zq = fmaf(z, z, zq);
    }
    zstat[a] = make_float2(zs * (1.0f / MC), zq * (1.0f / MC));
    float term = -__logf(s) + 0.5f * (s * s + l * l) - 0.5f;
    float w = wave_sum32(term);
    if ((threadIdx.x & 31) == 0) atomicAdd(&acc[0], (double)w);
}

// ---------------------------------------------------------------------------
// k_main: per wave, 16 observations x 4 ops.
//  Layer 1 (10->32) as two V_WMMA_F32_16X16X32_F16 tiles, A fragment built
//  in-register per the CDNA5 16-bit A layout:
//   lanes 0-15  : row M=lane,   V0-V3 = K0..7  (loc,sig,I,SigI,meta0..3)
//   lanes 0-15  : V4 = feats[8..9] (wavelength, 1/d^2); lanes 16-31 : all 0
//  B tiles: lanes 0-15 column N=lane hold K0..15 (W1 rows, zero-padded);
//  lanes 16-31 hold K16..31 (all zero since K=10).
//  Layer 2 (32->1): per-lane FMA + 16-lane xor reduce; softplus; scale
//  distributed to owner lanes through a tiny LDS slot.
//  MC statistics folded analytically via (E[z], E[z^2]) gathers.
// ---------------------------------------------------------------------------
__global__ void __launch_bounds__(256)
k_main(const int*    __restrict__ refl_ids,
       const float*  __restrict__ I,
       const float*  __restrict__ SigI,
       const int*    __restrict__ image_id,
       const float4* __restrict__ metadata4,
       const float*  __restrict__ wavelength,
       const float*  __restrict__ dHKL,
       const float2* __restrict__ as_tab,
       const float2* __restrict__ zstat,
       const float*  __restrict__ W1,
       const float*  __restrict__ b1,
       const float*  __restrict__ W2,
       const float*  __restrict__ b2,
       float* __restrict__ ll_img,
       float* __restrict__ ipred_mean) {           // [N_OPS][N_OBS]
    __shared__ float sbc[8][16];                    // per-wave scale slots

    const int lane = threadIdx.x & 31;
    const int wv   = threadIdx.x >> 5;
    const int wave = (blockIdx.x * blockDim.x + threadIdx.x) >> 5;
    const int base = wave * 16;
    if (base >= N_OBS) return;                 // wave-uniform exit (EXEC stays full)
    const int  col = lane & 15;
    const bool lo  = lane < 16;

    // B fragments (W1 tiles), biases, layer-2 weights
    v16h vb0 = {}; v16h vb1 = {};
#pragma unroll
    for (int k = 0; k < 10; ++k) {
        vb0[k] = lo ? (_Float16)W1[k * 32 + col]      : (_Float16)0.0f;
        vb1[k] = lo ? (_Float16)W1[k * 32 + 16 + col] : (_Float16)0.0f;
    }
    const float bias0 = b1[col], bias1 = b1[16 + col];
    const float w2a = W2[col], w2b = W2[16 + col];
    const float b2v = b2[0];

    // per-obs data, owned by lanes 0..15 (obs = base + col)
    const int n = base + col;
    float fI = 0.0f, fS = 1.0f, lsig = 0.0f, I2 = 0.0f, invs2 = 1.0f;
    int   img = 0;
    float fsh[8] = {0, 0, 0, 0, 0, 0, 0, 0};   // feats[2..9]
    if (lo) {
        fI   = I[n];
        fS   = SigI[n];
        lsig = __logf(fS);
        I2   = fI * fI;
        invs2 = 1.0f / (fS * fS);
        img  = image_id[n];
        float4 m4 = metadata4[n];
        fsh[0] = fI;   fsh[1] = fS;
        fsh[2] = m4.x; fsh[3] = m4.y; fsh[4] = m4.z; fsh[5] = m4.w;
        fsh[6] = wavelength[n];
        float d = dHKL[n];
        fsh[7] = 1.0f / (d * d);
    }
    // lanes 16-31 carry feats[8..9] of row (lane-16); constant across ops
    const float f8o = __shfl(fsh[6], col);
    const float f9o = __shfl(fsh[7], col);

    // shared (op-invariant) part of the A fragment
    v16h va_base = {};
    if (lo) {
#pragma unroll
        for (int j = 0; j < 8; ++j) va_base[2 + j] = (_Float16)fsh[j];
    }

#pragma unroll
    for (int op = 0; op < N_OPS; ++op) {
        float lv = 0.0f, sv = 0.0f, zm = 0.0f, zq = 0.0f;
        if (lo) {
            int rid = refl_ids[(size_t)op * N_OBS + n];
            float2 as = as_tab[rid];            // 8B L2-resident gather
            float2 zs = zstat[rid];             // 8B L2-resident gather
            lv = as.x; sv = as.y;
            zm = zs.x; zq = zs.y;
        }
        v16h va = va_base;
        if (lo) {
            va[0] = (_Float16)lv;
            va[1] = (_Float16)sv;
        }
        v8f c0 = {}; v8f c1 = {};
        c0 = __builtin_amdgcn_wmma_f32_16x16x32_f16(false, va, false, vb0,
                                                    (short)0, c0, false, false);
        c1 = __builtin_amdgcn_wmma_f32_16x16x32_f16(false, va, false, vb1,
                                                    (short)0, c1, false, false);
        // layer 2: relu + dot(32) via per-lane FMA + 16-lane xor reduction
        float sc[8];
#pragma unroll
        for (int r = 0; r < 8; ++r) {
            float h0 = fmaxf(c0[r] + bias0, 0.0f);
            float h1 = fmaxf(c1[r] + bias1, 0.0f);
            float p = fmaf(h1, w2b, h0 * w2a);
            p += __shfl_xor(p, 1);
            p += __shfl_xor(p, 2);
            p += __shfl_xor(p, 4);
            p += __shfl_xor(p, 8);
            sc[r] = softplus_f(p + b2v);       // lanes 0-15: obs r; 16-31: obs r+8
        }
        // route scale to owner lanes through LDS (contiguous, merges to b128)
        if (lane == 0) {
#pragma unroll
            for (int r = 0; r < 8; ++r) sbc[wv][r] = sc[r];
        }
        if (lane == 16) {
#pragma unroll
            for (int r = 0; r < 8; ++r) sbc[wv][8 + r] = sc[r];
        }
        if (lo) {
            float scale = sbc[wv][col];
            // exact MC-mean via moments:
            //   mean(ll) = -0.5/s^2 (I^2 - 2 I scale E[z] + scale^2 E[z^2]) - ln s - c
            float ipm = zm * scale;
            float q = fmaf(scale * scale, zq, fmaf(-2.0f * fI * scale, zm, I2));
            float llm = fmaf(-0.5f * invs2, q, -lsig - 0.5f * LOG_2PI);
            atomicAdd(&ll_img[img * N_OPS + op], llm);
            ipred_mean[(size_t)op * N_OBS + n] = ipm;
        }
    }
}

// ---------------------------------------------------------------------------
// k_argmax: per image, pick op with max mean-LL (first-max tie like jnp.argmax),
// accumulate sum of maxima for ll term, emit op_idx.
// ---------------------------------------------------------------------------
__global__ void k_argmax(const float* __restrict__ ll_img,
                         int* __restrict__ op_idx,
                         float* __restrict__ d_out,
                         double* acc) {
    int i = blockIdx.x * blockDim.x + threadIdx.x;   // 8192 divisible by 256
    float best = ll_img[i * N_OPS];
    int   bi = 0;
#pragma unroll
    for (int op = 1; op < N_OPS; ++op) {
        float v = ll_img[i * N_OPS + op];
        if (v > best) { best = v; bi = op; }
    }
    op_idx[i] = bi;
    d_out[2 + i] = (float)bi;
    float s = wave_sum32(best);
    if ((threadIdx.x & 31) == 0) atomicAdd(&acc[1], (double)s);
}

// ---------------------------------------------------------------------------
// k_cc: single-pass weighted moments for Pearson (f64 atomics, CDNA5 native)
// ---------------------------------------------------------------------------
__global__ void k_cc(const float* __restrict__ I,
                     const float* __restrict__ SigI,
                     const int*   __restrict__ image_id,
                     const int*   __restrict__ op_idx,
                     const float* __restrict__ ipred_mean,   // [N_OPS][N_OBS]
                     double* acc) {
    int nn = blockIdx.x * blockDim.x + threadIdx.x;
    float w = 0.0f, x = 0.0f, y = 0.0f;
    if (nn < N_OBS) {
        float s = SigI[nn];
        w = 1.0f / (s * s);
        x = I[nn];
        int op = op_idx[image_id[nn]];
        y = ipred_mean[(size_t)op * N_OBS + nn];
    }
    float sw  = wave_sum32(w);
    float sx  = wave_sum32(w * x);
    float sy  = wave_sum32(w * y);
    float sxx = wave_sum32(w * x * x);
    float syy = wave_sum32(w * y * y);
    float sxy = wave_sum32(w * x * y);
    if ((threadIdx.x & 31) == 0) {
        atomicAdd(&acc[2], (double)sw);
        atomicAdd(&acc[3], (double)sx);
        atomicAdd(&acc[4], (double)sy);
        atomicAdd(&acc[5], (double)sxx);
        atomicAdd(&acc[6], (double)syy);
        atomicAdd(&acc[7], (double)sxy);
    }
}

// ---------------------------------------------------------------------------
// k_fin: combine scalars -> elbo, cc
// ---------------------------------------------------------------------------
__global__ void k_fin(const double* acc, float* d_out) {
    if (threadIdx.x == 0 && blockIdx.x == 0) {
        double kl = acc[0] / (double)N_ASU;
        double ll = acc[1] / (double)N_IMAGES;
        double W = acc[2], Sx = acc[3], Sy = acc[4];
        double Sxx = acc[5], Syy = acc[6], Sxy = acc[7];
        double mx = Sx / W, my = Sy / W;
        double cxy = Sxy / W - mx * my;
        double cx  = Sxx / W - mx * mx;
        double cy  = Syy / W - my * my;
        d_out[0] = (float)(-ll + kl);
        d_out[1] = (float)(cxy / sqrt(cx * cy));
    }
}

extern "C" void kernel_launch(void* const* d_in, const int* in_sizes, int n_in,
                              void* d_out, int out_size, void* d_ws, size_t ws_size,
                              hipStream_t stream) {
    const int*   refl_ids   = (const int*)  d_in[0];
    const float* I          = (const float*)d_in[1];
    const float* SigI       = (const float*)d_in[2];
    const int*   image_id   = (const int*)  d_in[3];
    const float* metadata   = (const float*)d_in[4];
    const float* wavelength = (const float*)d_in[5];
    const float* dHKL       = (const float*)d_in[6];
    const float* loc        = (const float*)d_in[7];
    const float* raw_scale  = (const float*)d_in[8];
    const float* eps        = (const float*)d_in[9];
    const float* W1         = (const float*)d_in[10];
    const float* b1         = (const float*)d_in[11];
    const float* W2         = (const float*)d_in[12];
    const float* b2         = (const float*)d_in[13];
    float* out = (float*)d_out;

    // workspace layout (~18.2 MB)
    char* ws = (char*)d_ws;
    size_t off = 0;
    float2* as_tab = (float2*)(ws + off); off += (size_t)N_ASU * 8;            // 1 MB
    float2* zstat  = (float2*)(ws + off); off += (size_t)N_ASU * 8;            // 1 MB
    float*  ipm    = (float*) (ws + off); off += (size_t)N_OPS * N_OBS * 4;    // 16 MB
    float*  ll_img = (float*) (ws + off); off += (size_t)N_IMAGES * N_OPS * 4; // 128 KB
    int*    opidx  = (int*)   (ws + off); off += (size_t)N_IMAGES * 4;         // 32 KB
    double* acc    = (double*)(ws + off);                                      // 8 f64

    k_zero<<<(N_IMAGES * N_OPS + 255) / 256, 256, 0, stream>>>(ll_img, acc);
    k_prep<<<N_ASU / 256, 256, 0, stream>>>(loc, raw_scale, eps, as_tab, zstat, acc);
    const int waves  = (N_OBS + 15) / 16;          // 62500 waves, 16 obs each
    const int blocks = (waves + 7) / 8;            // 8 waves / 256-thread block
    k_main<<<blocks, 256, 0, stream>>>(refl_ids, I, SigI, image_id,
                                       (const float4*)metadata,
                                       wavelength, dHKL, as_tab, zstat,
                                       W1, b1, W2, b2, ll_img, ipm);
    k_argmax<<<N_IMAGES / 256, 256, 0, stream>>>(ll_img, opidx, out, acc);
    k_cc<<<(N_OBS + 255) / 256, 256, 0, stream>>>(I, SigI, image_id, opidx, ipm, acc);
    k_fin<<<1, 1, 0, stream>>>(acc, out);
}
